// ResidualizeEMA_22548578304162
// MI455X (gfx1250) — compile-verified, matrix operand order checked by reading, not run on hardware
//
#include <hip/hip_runtime.h>

typedef __attribute__((ext_vector_type(16))) __bf16 v16bf;
typedef __attribute__((ext_vector_type(8)))  __bf16 v8bf;
typedef __attribute__((ext_vector_type(8)))  float  v8f;
typedef __attribute__((ext_vector_type(4)))  float  v4f;
typedef __attribute__((ext_vector_type(4)))  unsigned int v4u;
typedef __attribute__((ext_vector_type(8)))  int    v8i;
typedef __attribute__((ext_vector_type(4)))  int    v4i;

#define ALPHA_F   0.2f
#define LOG2_1MA  (-0.32192809488736234f)   // log2(0.8)
#define B_  64
#define T_  2048
#define F_  256
#define M_  (B_ * T_)

// ---------------------------------------------------------------------------
// Kernel 1a: partial EMA sums. grid=(64,16), block=256.
// ---------------------------------------------------------------------------
__global__ __launch_bounds__(256) void ema_partial(const float* __restrict__ x,
                                                   float* __restrict__ partial) {
  const int b = blockIdx.x;
  const int chunk = blockIdx.y;           // 0..15
  const int f = threadIdx.x;              // 0..255
  const int TC = T_ / 16;                 // 128
  const int t0 = chunk * TC;
  const float* px = x + ((size_t)b * T_ + t0) * F_ + f;
  float acc = 0.0f;
  for (int i = 0; i < TC; ++i) {
    int t = t0 + i;
    float w = ALPHA_F * exp2f((float)(T_ - 1 - t) * LOG2_1MA);
    if (t == 0) w = exp2f((float)(T_ - 1) * LOG2_1MA);   // w[0] = (1-a)^(T-1)
    acc = fmaf(w, px[(size_t)i * F_], acc);
  }
  partial[((size_t)b * 16 + chunk) * F_ + f] = acc;
}

// ---------------------------------------------------------------------------
// Kernel 1b: addv[b][f] = bias[f] + sum_c partial[b][c][f].  grid=64, block=256
// ---------------------------------------------------------------------------
__global__ __launch_bounds__(256) void ema_reduce(const float* __restrict__ partial,
                                                  const float* __restrict__ bias,
                                                  float* __restrict__ addv) {
  const int b = blockIdx.x;
  const int f = threadIdx.x;
  float s = bias[f];
  for (int c = 0; c < 16; ++c) s += partial[((size_t)b * 16 + c) * F_ + f];
  addv[(size_t)b * F_ + f] = s;
}

// ---------------------------------------------------------------------------
// Kernel 1c: one-time W split+transpose: whiT/wloT[g][f] bf16. grid=64, blk=256
// ---------------------------------------------------------------------------
__global__ __launch_bounds__(256) void wsplit(const float* __restrict__ W,
                                              __bf16* __restrict__ whiT,
                                              __bf16* __restrict__ wloT) {
  int idx = blockIdx.x * 1024 + threadIdx.x * 4;
  int f = idx >> 8;
  int g0 = idx & 255;
  v4f wv = *(const v4f*)(W + idx);       // coalesced read of W[f][g0..g0+3]
#pragma unroll
  for (int q = 0; q < 4; ++q) {
    float v = wv[q];
    __bf16 h = (__bf16)v;
    whiT[(size_t)(g0 + q) * F_ + f] = h;
    wloT[(size_t)(g0 + q) * F_ + f] = (__bf16)(v - (float)h);
  }
}

// ---------------------------------------------------------------------------
// Kernel 2: bf16 split-fp32 WMMA GEMM; W^T hi/lo planes DMA'd into LDS by TDM.
// LDS layout: column-major with +8 bf16 pad/row (inserted by TDM pad feature).
// ---------------------------------------------------------------------------
#define LDS_STRIDE 264   // 256 + 8 bf16 pad per column

// Issue one TENSOR_LOAD_TO_LDS: 256 rows x 128 dwords contiguous from `src`,
// +4 dwords of LDS padding after every 128 dwords (=> stride 132 dwords).
__device__ __forceinline__ void tdm_load_w(const __bf16* src, __bf16* lds_dst) {
  unsigned lds_off = (unsigned)(unsigned long long)(uintptr_t)lds_dst; // low 32 = LDS byte offset
  unsigned long long ga = (unsigned long long)(uintptr_t)src;
  v4u g0 = { 1u,                                        // count=1, load, no gather
             lds_off,                                   // D# lds_addr
             (unsigned)ga,                              // global_addr[31:0]
             (unsigned)((ga >> 32) & 0x1FFFFFFull) | (2u << 30) }; // addr[56:32] | type=2
  v8i g1 = { (int)0x07920000,        // mask=0 | data_size=4B | pad_en | interval=128dw | amount=4dw
             (int)(128u << 16),      // tensor_dim0 = 128 dwords (row of bf16 plane)
             (int)(256u << 16),      // tensor_dim1 = 256 rows
             (int)(128u << 16),      // tile_dim0  = 128 dwords
             (int)256,               // tile_dim1  = 256 (tile_dim2 = 0)
             (int)128,               // tensor_dim0_stride = 128 dwords
             0, 0 };                 // tensor_dim1_stride = 0 (unused, 2D)
  v4i gz4 = {0, 0, 0, 0};
  v8i gz8 = {0, 0, 0, 0, 0, 0, 0, 0};
  __builtin_amdgcn_tensor_load_to_lds(g0, g1, gz4, gz4, gz8, 0);
}

__global__ __launch_bounds__(256) void gemm_wmma(const float* __restrict__ x,
                                                 const __bf16* __restrict__ whiT,
                                                 const __bf16* __restrict__ wloT,
                                                 const float* __restrict__ addv,
                                                 float* __restrict__ out,
                                                 int rows_per_wg) {
  __shared__ __bf16 sWhi[F_ * LDS_STRIDE];
  __shared__ __bf16 sWlo[F_ * LDS_STRIDE];

  const int tid = threadIdx.x;

  // --- TDM: DMA both W planes (global -> LDS, padded) from wave 0 ---------
  if (tid < 32) {
    tdm_load_w(whiT, sWhi);
    tdm_load_w(wloT, sWlo);
    __builtin_amdgcn_s_wait_tensorcnt(0);
  }
  __syncthreads();

  const int wave = tid >> 5;
  const int lane = tid & 31;
  const int ln16 = lane & 15;
  const int half = lane >> 4;     // 0: K 0..7/16..23, 1: K 8..15/24..31
  const int kg = half * 8;

  const size_t chunk_start = (size_t)blockIdx.x * (size_t)rows_per_wg;
  const v8f vzero = {0.f, 0.f, 0.f, 0.f, 0.f, 0.f, 0.f, 0.f};

  for (int it = 0; it < rows_per_wg; it += 128) {
    const size_t m0 = chunk_start + (size_t)it + (size_t)wave * 16;
    const int bidx = (int)(m0 >> 11);                 // batch = row / T
    const float* pa = x + (m0 + (size_t)ln16) * F_ + kg;  // A-fragment base

    v8f acc[16];
#pragma unroll
    for (int j = 0; j < 16; ++j) acc[j] = vzero;

    for (int k8 = 0; k8 < 8; ++k8) {        // K steps of 32
      const int k0 = k8 * 32;
      // A fragment: 16 f32 per lane in ISA layout (two 8-wide K chunks)
      v4f a0 = *(const v4f*)(pa + k0);
      v4f a1 = *(const v4f*)(pa + k0 + 4);
      v4f a2 = *(const v4f*)(pa + k0 + 16);
      v4f a3 = *(const v4f*)(pa + k0 + 20);
      v16bf ahi, alo;
#pragma unroll
      for (int q = 0; q < 4; ++q) {
        float fv;
        fv = a0[q]; { __bf16 h = (__bf16)fv; ahi[q]      = h; alo[q]      = (__bf16)(fv - (float)h); }
        fv = a1[q]; { __bf16 h = (__bf16)fv; ahi[4 + q]  = h; alo[4 + q]  = (__bf16)(fv - (float)h); }
        fv = a2[q]; { __bf16 h = (__bf16)fv; ahi[8 + q]  = h; alo[8 + q]  = (__bf16)(fv - (float)h); }
        fv = a3[q]; { __bf16 h = (__bf16)fv; ahi[12 + q] = h; alo[12 + q] = (__bf16)(fv - (float)h); }
      }

      const int kb = k0 + kg;
#pragma unroll
      for (int j = 0; j < 16; ++j) {
        const __bf16* bh = &sWhi[(size_t)(j * 16 + ln16) * LDS_STRIDE + kb];
        const __bf16* bl = &sWlo[(size_t)(j * 16 + ln16) * LDS_STRIDE + kb];
        v8bf bh0 = *(const v8bf*)(bh);
        v8bf bh1 = *(const v8bf*)(bh + 16);
        v8bf bl0 = *(const v8bf*)(bl);
        v8bf bl1 = *(const v8bf*)(bl + 16);
        v16bf Bhi = __builtin_shufflevector(bh0, bh1, 0,1,2,3,4,5,6,7,8,9,10,11,12,13,14,15);
        v16bf Blo = __builtin_shufflevector(bl0, bl1, 0,1,2,3,4,5,6,7,8,9,10,11,12,13,14,15);
        // split-fp32: hi*hi + hi*lo + lo*hi, f32 accumulate
        acc[j] = __builtin_amdgcn_wmma_f32_16x16x32_bf16(false, ahi, false, Bhi,
                                                         (short)0, acc[j], false, false);
        acc[j] = __builtin_amdgcn_wmma_f32_16x16x32_bf16(false, ahi, false, Blo,
                                                         (short)0, acc[j], false, false);
        acc[j] = __builtin_amdgcn_wmma_f32_16x16x32_bf16(false, alo, false, Bhi,
                                                         (short)0, acc[j], false, false);
      }
    }

    // --- store: D layout => lane(<16): rows m0+r, lane(>=16): rows m0+8+r --
    const size_t row0 = m0 + (size_t)half * 8;
    const float* av = addv + (size_t)bidx * F_;
#pragma unroll
    for (int j = 0; j < 16; ++j) {
      const int n = j * 16 + ln16;
      const float a = av[n];
      float* po = out + row0 * F_ + n;
#pragma unroll
      for (int r = 0; r < 8; ++r) {
        po[(size_t)r * F_] = acc[j][r] + a;
      }
    }
  }
}

// ---------------------------------------------------------------------------
extern "C" void kernel_launch(void* const* d_in, const int* in_sizes, int n_in,
                              void* d_out, int out_size, void* d_ws, size_t ws_size,
                              hipStream_t stream) {
  const float* x    = (const float*)d_in[0];   // [64,2048,256] f32
  const float* W    = (const float*)d_in[1];   // [256,256] f32
  const float* bias = (const float*)d_in[2];   // [256] f32
  float* out = (float*)d_out;                  // [64,2048,256] f32

  float*  partial = (float*)d_ws;                       // 64*16*256 f32 = 1 MiB
  float*  addv    = partial + (size_t)B_ * 16 * F_;     // 64*256 f32
  __bf16* whiT    = (__bf16*)(addv + (size_t)B_ * F_);  // 256*256 bf16
  __bf16* wloT    = whiT + (size_t)F_ * F_;             // 256*256 bf16

  wsplit<<<dim3(64), dim3(256), 0, stream>>>(W, whiT, wloT);
  ema_partial<<<dim3(B_, 16), dim3(256), 0, stream>>>(x, partial);
  ema_reduce<<<dim3(B_), dim3(256), 0, stream>>>(partial, bias, addv);

  const int n_wg = 512;                        // 256 rows per WG, 2 iterations
  const int rows_per_wg = M_ / n_wg;
  gemm_wmma<<<dim3(n_wg), dim3(256), 0, stream>>>(x, whiT, wloT, addv, out, rows_per_wg);
}
